// FlaxDbrxBlock_29472065585703
// MI455X (gfx1250) — matrix operand order repaired and checked by reading, not compiled, not run on hardware
//
#include <hip/hip_runtime.h>
#include <hip/hip_bf16.h>
#include <cstring>

#define S_    1024
#define D_    2048
#define H_    16
#define KVH_  4
#define HD_   128
#define E_    8
#define F_    2048
#define NQKV_ 3072   // D + 2*KVH*HD

typedef __bf16 bf16;
typedef __attribute__((ext_vector_type(16))) __bf16 bf16x16;
typedef __attribute__((ext_vector_type(4)))  __bf16 bf16x4;
typedef __attribute__((ext_vector_type(8)))  float  f32x8;
typedef __attribute__((ext_vector_type(4)))  unsigned int uint4v;
typedef __attribute__((ext_vector_type(4)))  int i32x4;

union FragU { unsigned int u[8]; uint4v q[2]; bf16x16 v; };

__device__ __forceinline__ f32x8 wmma_bf16(const FragU& a, const FragU& b, f32x8 c) {
    return __builtin_amdgcn_wmma_f32_16x16x32_bf16(false, a.v, false, b.v, (short)0, c,
                                                   false, false);
}

// --- gfx1250 async global->LDS path (ASYNCcnt), with sync fallback ---------
#if defined(__gfx1250__) && __has_builtin(__builtin_amdgcn_global_load_async_to_lds_b128) && \
    __has_builtin(__builtin_amdgcn_s_wait_asynccnt)
#define ASYNC_LDS 1
#else
#define ASYNC_LDS 0
#endif

__device__ __forceinline__ void ld16_to_lds(const bf16* g, unsigned int* l) {
#if ASYNC_LDS
    __builtin_amdgcn_global_load_async_to_lds_b128(
        (__attribute__((address_space(1))) i32x4*)g,
        (__attribute__((address_space(3))) i32x4*)l, 0, 0);
#else
    *(uint4v*)l = *(const uint4v*)g;
#endif
}

template<int N>
__device__ __forceinline__ void wait_async() {
#if ASYNC_LDS
    __builtin_amdgcn_s_wait_asynccnt(N);
#endif
}

// ---------------------------------------------------------------------------
// Generic 128x128-tile GEMM:  C[M,N] = A[M,K](bf16, row-major) * B[N,K]^T(bf16)
// 8 wave32 waves (4x2); each wave owns 32x64 = 2x4 WMMA 16x16 tiles, K-step 32.
// Double-buffered LDS tiles staged with async global->LDS loads; fragments are
// read from LDS as 2x b128 per operand (40-bf16 pitch avoids bank conflicts).
// ---------------------------------------------------------------------------
enum : int { EPI_QKV = 0, EPI_SCORES = 1, EPI_BF16 = 2, EPI_ADDRES = 3, EPI_F32 = 4 };

struct GemmArgs {
    const bf16* A; long lda; long batchA; int bdivA;
    const bf16* B; long ldb; long batchB; int bdivB;
    int K;
    float* Cf; bf16* Cb; long ldc; long batchC; int bdivC;
    const float* add;      // EPI_ADDRES: residual input
    const int*   amask;    // EPI_SCORES: attention_mask[S]
    const int*   row_cnt;  // optional per-batch(z) valid-row count (early exit)
    float scale;           // EPI_SCORES
};

template<int EPI>
__global__ __launch_bounds__(256) void gemm128(GemmArgs g) {
    const int z       = blockIdx.z;
    const int rowBase = blockIdx.x * 128;
    const int colBase = blockIdx.y * 128;
    if (g.row_cnt && rowBase >= g.row_cnt[z]) return;

    if constexpr (EPI == EPI_SCORES) {
        // tile entirely above the causal diagonal: fill -FLT_MAX, skip GEMM
        if (colBase > rowBase + 127) {
            float* C = g.Cf + (long)z * g.batchC;
            for (int i = threadIdx.x; i < 128 * 128; i += 256) {
                int r = i >> 7, c = i & 127;
                C[(long)(rowBase + r) * g.ldc + colBase + c] = -3.4028235e38f;
            }
            return;
        }
    }

    const bf16* A = g.A + (long)(z / g.bdivA) * g.batchA;
    const bf16* B = g.B + (long)(z / g.bdivB) * g.batchB;

    __shared__ __align__(16) unsigned int As[2][128 * 20];
    __shared__ __align__(16) unsigned int Bs[2][128 * 20];

    const int tid  = threadIdx.x;
    const int lane = tid & 31;
    const int wv   = tid >> 5;
    const int wm   = wv & 3, wn = wv >> 2;
    const int hi   = lane >> 4, lr = lane & 15;

    // staging geometry: 512 16B units per tile, 2 per thread
    int sr[2], su[2];
    const bf16* Arow[2];
    const bf16* Brow[2];
    #pragma unroll
    for (int t = 0; t < 2; t++) {
        int i = tid + t * 256;
        sr[t] = i >> 2;
        su[t] = i & 3;
        Arow[t] = A + (long)(rowBase + sr[t]) * g.lda + su[t] * 8;
        Brow[t] = B + (long)(colBase + sr[t]) * g.ldb + su[t] * 8;
    }

    f32x8 acc[2][4];
    for (int a = 0; a < 2; a++)
        for (int b = 0; b < 4; b++)
            acc[a][b] = (f32x8){0.f,0.f,0.f,0.f,0.f,0.f,0.f,0.f};

    const int nsteps = g.K >> 5;

    // prologue: stage tile 0
    #pragma unroll
    for (int t = 0; t < 2; t++) {
        ld16_to_lds(Arow[t], &As[0][sr[t] * 20 + su[t] * 4]);
        ld16_to_lds(Brow[t], &Bs[0][sr[t] * 20 + su[t] * 4]);
    }

    for (int s = 0; s < nsteps; s++) {
        const int buf = s & 1;
        if (s + 1 < nsteps) {
            const int k1 = (s + 1) << 5;
            #pragma unroll
            for (int t = 0; t < 2; t++) {
                ld16_to_lds(Arow[t] + k1, &As[buf ^ 1][sr[t] * 20 + su[t] * 4]);
                ld16_to_lds(Brow[t] + k1, &Bs[buf ^ 1][sr[t] * 20 + su[t] * 4]);
            }
            wait_async<4>();   // current tile's 4 loads done; next tile in flight
        } else {
            wait_async<0>();
        }
        __syncthreads();

        FragU bfr[4];
        #pragma unroll
        for (int tn = 0; tn < 4; tn++) {
            const unsigned int* p = &Bs[buf][(wn * 64 + tn * 16 + lr) * 20 + hi * 8];
            bfr[tn].q[0] = *(const uint4v*)p;
            bfr[tn].q[1] = *(const uint4v*)(p + 4);
        }
        #pragma unroll
        for (int tm = 0; tm < 2; tm++) {
            FragU af;
            const unsigned int* pa = &As[buf][(wm * 32 + tm * 16 + lr) * 20 + hi * 8];
            af.q[0] = *(const uint4v*)pa;
            af.q[1] = *(const uint4v*)(pa + 4);
            #pragma unroll
            for (int tn = 0; tn < 4; tn++)
                acc[tm][tn] = wmma_bf16(af, bfr[tn], acc[tm][tn]);
        }
        __syncthreads();
    }

    // epilogue — C f32 layout: lanes 0-15: M=r, N=lane; lanes 16-31: M=8+r, N=lane-16
    #pragma unroll
    for (int tm = 0; tm < 2; tm++)
    #pragma unroll
    for (int tn = 0; tn < 4; tn++) {
        const int col = colBase + wn * 64 + tn * 16 + lr;
        #pragma unroll
        for (int r8 = 0; r8 < 8; r8++) {
            const int row = rowBase + wm * 32 + tm * 16 + hi * 8 + r8;
            float v = acc[tm][tn][r8];
            if constexpr (EPI == EPI_QKV) {
                g.Cf[(long)row * g.ldc + col] = fminf(fmaxf(v, -8.0f), 8.0f);
            } else if constexpr (EPI == EPI_SCORES) {
                float o = v * g.scale;
                if (col > row || g.amask[col] == 0) o = -3.4028235e38f;
                (g.Cf + (long)z * g.batchC)[(long)row * g.ldc + col] = o;
            } else if constexpr (EPI == EPI_BF16) {
                (g.Cb + (long)(z / g.bdivC) * g.batchC)[(long)row * g.ldc + col] = (bf16)v;
            } else if constexpr (EPI == EPI_ADDRES) {
                g.Cf[(long)row * g.ldc + col] = v + g.add[(long)row * g.ldc + col];
            } else { // EPI_F32
                (g.Cf + (long)z * g.batchC)[(long)row * g.ldc + col] = v;
            }
        }
    }
}

// ---------------------------------------------------------------------------
// MoE first stage, fused dual-B: mid = bf16( silu(h2*W1^T) * (h2*V1^T) )
// A rows gathered via per-expert token list. Double-buffered async staging.
// ---------------------------------------------------------------------------
__global__ __launch_bounds__(256) void moe1_kernel(const bf16* __restrict__ h2,
                                                   const bf16* __restrict__ w1b,
                                                   const bf16* __restrict__ v1b,
                                                   const int*  __restrict__ tlist,
                                                   const int*  __restrict__ cnt,
                                                   bf16* __restrict__ mid) {
    const int e       = blockIdx.z;
    const int c       = cnt[e];
    const int rowBase = blockIdx.x * 128;
    const int colBase = blockIdx.y * 128;
    if (rowBase >= c) return;
    const int*  ridx = tlist + e * S_;
    const bf16* B1   = w1b + (long)e * F_ * D_;
    const bf16* B2   = v1b + (long)e * F_ * D_;
    bf16*       M    = mid + (long)e * S_ * F_;

    __shared__ __align__(16) unsigned int As[2][128 * 20];
    __shared__ __align__(16) unsigned int B1s[2][128 * 20];
    __shared__ __align__(16) unsigned int B2s[2][128 * 20];

    const int tid  = threadIdx.x;
    const int lane = tid & 31, wvid = tid >> 5;
    const int wm = wvid & 3, wn = wvid >> 2;
    const int hi = lane >> 4, lr = lane & 15;

    int sr[2], su[2];
    const bf16* Arow[2];
    const bf16* B1row[2];
    const bf16* B2row[2];
    #pragma unroll
    for (int t = 0; t < 2; t++) {
        int i  = tid + t * 256;
        sr[t] = i >> 2;
        su[t] = i & 3;
        int rr = rowBase + sr[t];
        int gr = (rr < c) ? ridx[rr] : ridx[0];
        Arow[t]  = h2 + (long)gr * D_ + su[t] * 8;
        B1row[t] = B1 + (long)(colBase + sr[t]) * D_ + su[t] * 8;
        B2row[t] = B2 + (long)(colBase + sr[t]) * D_ + su[t] * 8;
    }

    f32x8 acc1[2][4], acc2[2][4];
    for (int a = 0; a < 2; a++)
        for (int b = 0; b < 4; b++) {
            acc1[a][b] = (f32x8){0.f,0.f,0.f,0.f,0.f,0.f,0.f,0.f};
            acc2[a][b] = (f32x8){0.f,0.f,0.f,0.f,0.f,0.f,0.f,0.f};
        }

    #pragma unroll
    for (int t = 0; t < 2; t++) {
        const int o = sr[t] * 20 + su[t] * 4;
        ld16_to_lds(Arow[t],  &As[0][o]);
        ld16_to_lds(B1row[t], &B1s[0][o]);
        ld16_to_lds(B2row[t], &B2s[0][o]);
    }

    const int nsteps = D_ >> 5;
    for (int s = 0; s < nsteps; s++) {
        const int buf = s & 1;
        if (s + 1 < nsteps) {
            const int k1 = (s + 1) << 5;
            #pragma unroll
            for (int t = 0; t < 2; t++) {
                const int o = sr[t] * 20 + su[t] * 4;
                ld16_to_lds(Arow[t] + k1,  &As[buf ^ 1][o]);
                ld16_to_lds(B1row[t] + k1, &B1s[buf ^ 1][o]);
                ld16_to_lds(B2row[t] + k1, &B2s[buf ^ 1][o]);
            }
            wait_async<6>();
        } else {
            wait_async<0>();
        }
        __syncthreads();

        FragU b1f[4], b2f[4];
        #pragma unroll
        for (int tn = 0; tn < 4; tn++) {
            const int o = (wn * 64 + tn * 16 + lr) * 20 + hi * 8;
            const unsigned int* p1 = &B1s[buf][o];
            const unsigned int* p2 = &B2s[buf][o];
            b1f[tn].q[0] = *(const uint4v*)p1; b1f[tn].q[1] = *(const uint4v*)(p1 + 4);
            b2f[tn].q[0] = *(const uint4v*)p2; b2f[tn].q[1] = *(const uint4v*)(p2 + 4);
        }
        #pragma unroll
        for (int tm = 0; tm < 2; tm++) {
            FragU af;
            const unsigned int* pa = &As[buf][(wm * 32 + tm * 16 + lr) * 20 + hi * 8];
            af.q[0] = *(const uint4v*)pa;
            af.q[1] = *(const uint4v*)(pa + 4);
            #pragma unroll
            for (int tn = 0; tn < 4; tn++) {
                acc1[tm][tn] = wmma_bf16(af, b1f[tn], acc1[tm][tn]);
                acc2[tm][tn] = wmma_bf16(af, b2f[tn], acc2[tm][tn]);
            }
        }
        __syncthreads();
    }

    #pragma unroll
    for (int tm = 0; tm < 2; tm++)
    #pragma unroll
    for (int tn = 0; tn < 4; tn++) {
        const int col = colBase + wn * 64 + tn * 16 + lr;
        #pragma unroll
        for (int r8 = 0; r8 < 8; r8++) {
            const int row = rowBase + wm * 32 + tm * 16 + hi * 8 + r8;
            float x1 = acc1[tm][tn][r8];
            float x2 = acc2[tm][tn][r8];
            float sv = x1 / (1.0f + expf(-x1));   // SiLU
            M[(long)row * F_ + col] = (bf16)(sv * x2);
        }
    }
}

// ---------------------------------------------------------------------------
// Elementwise / reduction kernels
// ---------------------------------------------------------------------------
__global__ __launch_bounds__(256) void ln1_kernel(const float* __restrict__ x,
                                                  const float* __restrict__ w,
                                                  bf16* __restrict__ out) {
    const int s = blockIdx.x;
    const float* xr = x + (long)s * D_;
    __shared__ float red[256];
    float loc[8];
    float sum = 0.f;
    #pragma unroll
    for (int i = 0; i < 8; i++) { loc[i] = xr[threadIdx.x + i * 256]; sum += loc[i]; }
    red[threadIdx.x] = sum; __syncthreads();
    for (int o = 128; o > 0; o >>= 1) {
        if (threadIdx.x < o) red[threadIdx.x] += red[threadIdx.x + o];
        __syncthreads();
    }
    float mu = red[0] * (1.0f / D_); __syncthreads();
    float s2 = 0.f;
    #pragma unroll
    for (int i = 0; i < 8; i++) { float d = loc[i] - mu; s2 += d * d; }
    red[threadIdx.x] = s2; __syncthreads();
    for (int o = 128; o > 0; o >>= 1) {
        if (threadIdx.x < o) red[threadIdx.x] += red[threadIdx.x + o];
        __syncthreads();
    }
    float rstd = rsqrtf(red[0] * (1.0f / D_) + 1e-5f);
    #pragma unroll
    for (int i = 0; i < 8; i++) {
        int d = threadIdx.x + i * 256;
        out[(long)s * D_ + d] = (bf16)((loc[i] - mu) * rstd * w[d]);
    }
}

__global__ __launch_bounds__(256) void ln2_router_kernel(const float* __restrict__ resid,
                                                         const float* __restrict__ w,
                                                         const float* __restrict__ rw,
                                                         bf16* __restrict__ h2b,
                                                         int* cnt, int* tlist,
                                                         int* tok_e, int* tok_slot,
                                                         float* tok_w) {
    const int s = blockIdx.x;
    const float* xr = resid + (long)s * D_;
    __shared__ float red[256];
    __shared__ float sl[E_];
    float loc[8];
    float sum = 0.f;
    #pragma unroll
    for (int i = 0; i < 8; i++) { loc[i] = xr[threadIdx.x + i * 256]; sum += loc[i]; }
    red[threadIdx.x] = sum; __syncthreads();
    for (int o = 128; o > 0; o >>= 1) {
        if (threadIdx.x < o) red[threadIdx.x] += red[threadIdx.x + o];
        __syncthreads();
    }
    float mu = red[0] * (1.0f / D_); __syncthreads();
    float s2 = 0.f;
    #pragma unroll
    for (int i = 0; i < 8; i++) { float d = loc[i] - mu; s2 += d * d; }
    red[threadIdx.x] = s2; __syncthreads();
    for (int o = 128; o > 0; o >>= 1) {
        if (threadIdx.x < o) red[threadIdx.x] += red[threadIdx.x + o];
        __syncthreads();
    }
    float rstd = rsqrtf(red[0] * (1.0f / D_) + 1e-5f);

    float lg[E_];
    #pragma unroll
    for (int e = 0; e < E_; e++) lg[e] = 0.f;
    #pragma unroll
    for (int i = 0; i < 8; i++) {
        int d = threadIdx.x + i * 256;
        float h = (loc[i] - mu) * rstd * w[d];
        h2b[(long)s * D_ + d] = (bf16)h;
        #pragma unroll
        for (int e = 0; e < E_; e++) lg[e] += h * rw[(long)d * E_ + e];
    }
    for (int e = 0; e < E_; e++) {
        __syncthreads();
        red[threadIdx.x] = lg[e]; __syncthreads();
        for (int o = 128; o > 0; o >>= 1) {
            if (threadIdx.x < o) red[threadIdx.x] += red[threadIdx.x + o];
            __syncthreads();
        }
        if (threadIdx.x == 0) sl[e] = red[0];
    }
    __syncthreads();
    if (threadIdx.x == 0) {
        float mx = sl[0];
        for (int e = 1; e < E_; e++) mx = fmaxf(mx, sl[e]);
        float wv[E_], tot = 0.f;
        for (int e = 0; e < E_; e++) { wv[e] = expf(sl[e] - mx); tot += wv[e]; }
        for (int e = 0; e < E_; e++) wv[e] /= tot;
        int e0 = 0;
        for (int e = 1; e < E_; e++) if (wv[e] > wv[e0]) e0 = e;
        int e1 = (e0 == 0) ? 1 : 0;
        for (int e = 0; e < E_; e++) if (e != e0 && wv[e] > wv[e1]) e1 = e;
        int s0 = atomicAdd(&cnt[e0], 1); tlist[e0 * S_ + s0] = s;
        int s1 = atomicAdd(&cnt[e1], 1); tlist[e1 * S_ + s1] = s;
        tok_e[2 * s] = e0;  tok_e[2 * s + 1] = e1;
        tok_slot[2 * s] = s0; tok_slot[2 * s + 1] = s1;
        float rn = wv[e0] + wv[e1];
        tok_w[2 * s] = wv[e0] / rn; tok_w[2 * s + 1] = wv[e1] / rn;
    }
}

__global__ __launch_bounds__(256) void rope_kernel(const float* __restrict__ qkv,
                                                   const int* __restrict__ pos,
                                                   bf16* __restrict__ qb,
                                                   bf16* __restrict__ kb,
                                                   bf16* __restrict__ vt) {
    const int s = blockIdx.x;
    const float p = (float)pos[s];
    const float* row = qkv + (long)s * NQKV_;
    const float lnb = logf(500000.0f);
    // Q: 16 heads * 64 rotation pairs
    #pragma unroll
    for (int i = 0; i < 4; i++) {
        int pr = threadIdx.x + i * 256;
        int h = pr >> 6, j = pr & 63;
        float inv = expf(-((float)j / 64.0f) * lnb);
        float fr = p * inv, c, sn;
        sincosf(fr, &sn, &c);
        float x1 = row[h * HD_ + j], x2 = row[h * HD_ + 64 + j];
        qb[(long)s * D_ + h * HD_ + j]      = (bf16)(x1 * c - x2 * sn);
        qb[(long)s * D_ + h * HD_ + 64 + j] = (bf16)(x2 * c + x1 * sn);
    }
    // K: 4 heads * 64 pairs
    {
        int pr = threadIdx.x;
        int h = pr >> 6, j = pr & 63;
        float inv = expf(-((float)j / 64.0f) * lnb);
        float fr = p * inv, c, sn;
        sincosf(fr, &sn, &c);
        const float* kr = row + D_;
        float x1 = kr[h * HD_ + j], x2 = kr[h * HD_ + 64 + j];
        kb[(long)s * (KVH_ * HD_) + h * HD_ + j]      = (bf16)(x1 * c - x2 * sn);
        kb[(long)s * (KVH_ * HD_) + h * HD_ + 64 + j] = (bf16)(x2 * c + x1 * sn);
    }
    // V: transpose-convert to (kvh, hd, s)
    #pragma unroll
    for (int i = 0; i < 2; i++) {
        int e = threadIdx.x + i * 256;
        int kvh = e >> 7, hd = e & 127;
        vt[((long)kvh * HD_ + hd) * S_ + s] = (bf16)row[D_ + KVH_ * HD_ + e];
    }
}

__global__ __launch_bounds__(256) void softmax_kernel(const float* __restrict__ sc,
                                                      bf16* __restrict__ pb) {
    const int s = blockIdx.x, h = blockIdx.y;
    const float* r = sc + ((long)h * S_ + s) * S_;
    bf16* o = pb + ((long)h * S_ + s) * S_;
    __shared__ float red[256];
    float loc[4];
    float mx = -3.4028235e38f;
    #pragma unroll
    for (int i = 0; i < 4; i++) { loc[i] = r[threadIdx.x + i * 256]; mx = fmaxf(mx, loc[i]); }
    red[threadIdx.x] = mx; __syncthreads();
    for (int off = 128; off > 0; off >>= 1) {
        if (threadIdx.x < off) red[threadIdx.x] = fmaxf(red[threadIdx.x], red[threadIdx.x + off]);
        __syncthreads();
    }
    mx = red[0]; __syncthreads();
    float sum = 0.f;
    #pragma unroll
    for (int i = 0; i < 4; i++) { loc[i] = expf(loc[i] - mx); sum += loc[i]; }
    red[threadIdx.x] = sum; __syncthreads();
    for (int off = 128; off > 0; off >>= 1) {
        if (threadIdx.x < off) red[threadIdx.x] += red[threadIdx.x + off];
        __syncthreads();
    }
    float inv = 1.0f / red[0];
    #pragma unroll
    for (int i = 0; i < 4; i++) o[threadIdx.x + i * 256] = (bf16)(loc[i] * inv);
}

// in: (batch, R, C) f32  ->  out: (batch, C, R) bf16
__global__ void transpose_bf16_kernel(const float* __restrict__ in,
                                      bf16* __restrict__ out, int R, int C) {
    __shared__ float t[32][33];
    const long b = blockIdx.z;
    const float* src = in + b * (long)R * C;
    bf16* dst = out + b * (long)R * C;
    const int c0 = blockIdx.x * 32, r0 = blockIdx.y * 32;
    for (int i = threadIdx.y; i < 32; i += 8)
        t[i][threadIdx.x] = src[(long)(r0 + i) * C + c0 + threadIdx.x];
    __syncthreads();
    for (int i = threadIdx.y; i < 32; i += 8)
        dst[(long)(c0 + i) * R + r0 + threadIdx.x] = (bf16)t[threadIdx.x][i];
}

__global__ void cvt_bf16_kernel(const float* __restrict__ in, bf16* __restrict__ out, long n) {
    long i = (long)blockIdx.x * 256 + threadIdx.x;
    const long stride = (long)gridDim.x * 256;
    const float4* in4 = (const float4*)in;
    bf16x4* out4 = (bf16x4*)out;
    const long n4 = n >> 2;
    for (; i < n4; i += stride) {
        float4 v = in4[i];
        bf16x4 o = {(bf16)v.x, (bf16)v.y, (bf16)v.z, (bf16)v.w};
        out4[i] = o;
    }
}

__global__ __launch_bounds__(256) void moe_final_kernel(float* __restrict__ out,
                                                        const float* __restrict__ ybuf,
                                                        const int* __restrict__ tok_e,
                                                        const int* __restrict__ tok_slot,
                                                        const float* __restrict__ tok_w) {
    const int s = blockIdx.x;
    const int d = blockIdx.y * 256 + threadIdx.x;
    const int e0 = tok_e[2 * s], e1 = tok_e[2 * s + 1];
    const int s0 = tok_slot[2 * s], s1 = tok_slot[2 * s + 1];
    const float w0 = tok_w[2 * s], w1v = tok_w[2 * s + 1];
    const long o = (long)s * D_ + d;
    float y0 = ybuf[((long)e0 * S_ + s0) * D_ + d];
    float y1 = ybuf[((long)e1 * S_ + s1) * D_ + d];
    out[o] = out[o] + w0 * y0 + w1v * y1;
}

// ---------------------------------------------------------------------------
extern "C" void kernel_launch(void* const* d_in, const int* in_sizes, int n_in,
                              void* d_out, int out_size, void* d_ws, size_t ws_size,
                              hipStream_t stream) {
    (void)in_sizes; (void)n_in; (void)out_size; (void)ws_size;
    const float* x     = (const float*)d_in[0];
    const int*   amask = (const int*)d_in[1];
    const int*   pos   = (const int*)d_in[2];
    const float* n1w   = (const float*)d_in[3];
    const float* n2w   = (const float*)d_in[4];
    const float* wqkv  = (const float*)d_in[5];
    const float* wout  = (const float*)d_in[6];
    const float* wrout = (const float*)d_in[7];
    const float* w1    = (const float*)d_in[8];
    const float* v1    = (const float*)d_in[9];
    const float* w2    = (const float*)d_in[10];
    float* out = (float*)d_out;

    char* base = (char*)d_ws;
    size_t off = 0;
    auto alloc = [&](size_t nbytes) -> void* {
        void* p = base + off;
        off += (nbytes + 255) & ~(size_t)255;
        return p;
    };
    bf16*  wqkvT  = (bf16*)alloc((size_t)NQKV_ * D_ * 2);
    bf16*  woutT  = (bf16*)alloc((size_t)D_ * D_ * 2);
    bf16*  w1b    = (bf16*)alloc((size_t)E_ * F_ * D_ * 2);
    bf16*  v1b    = (bf16*)alloc((size_t)E_ * F_ * D_ * 2);
    bf16*  w2T    = (bf16*)alloc((size_t)E_ * D_ * F_ * 2);
    bf16*  h1b    = (bf16*)alloc((size_t)S_ * D_ * 2);
    float* qkvf   = (float*)alloc((size_t)S_ * NQKV_ * 4);
    bf16*  qb     = (bf16*)alloc((size_t)S_ * D_ * 2);
    bf16*  kb     = (bf16*)alloc((size_t)S_ * KVH_ * HD_ * 2);
    bf16*  vtb    = (bf16*)alloc((size_t)KVH_ * HD_ * S_ * 2);
    float* scores = (float*)alloc((size_t)H_ * S_ * S_ * 4);
    bf16*  pb     = (bf16*)alloc((size_t)H_ * S_ * S_ * 2);
    bf16*  attnb  = (bf16*)alloc((size_t)S_ * D_ * 2);
    bf16*  h2b    = (bf16*)alloc((size_t)S_ * D_ * 2);
    bf16*  midb   = (bf16*)alloc((size_t)E_ * S_ * F_ * 2);
    float* ybuf   = (float*)alloc((size_t)E_ * S_ * D_ * 4);
    int*   cnt    = (int*)alloc(E_ * sizeof(int));
    int*   tlist  = (int*)alloc((size_t)E_ * S_ * sizeof(int));
    int*   tok_e  = (int*)alloc((size_t)S_ * 2 * sizeof(int));
    int*   tok_sl = (int*)alloc((size_t)S_ * 2 * sizeof(int));
    float* tok_w  = (float*)alloc((size_t)S_ * 2 * sizeof(float));

    // ---- weight repack: fp32 -> bf16 (+ transpose so B is always (N,K)) ----
    dim3 tb(32, 8);
    transpose_bf16_kernel<<<dim3(NQKV_ / 32, D_ / 32, 1), tb, 0, stream>>>(wqkv, wqkvT, D_, NQKV_);
    transpose_bf16_kernel<<<dim3(D_ / 32, D_ / 32, 1), tb, 0, stream>>>(wout, woutT, D_, D_);
    transpose_bf16_kernel<<<dim3(D_ / 32, F_ / 32, E_), tb, 0, stream>>>(w2, w2T, F_, D_);
    cvt_bf16_kernel<<<2048, 256, 0, stream>>>(w1, w1b, (long)E_ * F_ * D_);
    cvt_bf16_kernel<<<2048, 256, 0, stream>>>(v1, v1b, (long)E_ * F_ * D_);

    // ---- LN1 -> QKV (clipped) -> RoPE ----
    ln1_kernel<<<S_, 256, 0, stream>>>(x, n1w, h1b);

    GemmArgs g;
    auto reset = [&]() { memset(&g, 0, sizeof(g)); g.bdivA = g.bdivB = g.bdivC = 1; g.scale = 1.f; };

    reset();
    g.A = h1b; g.lda = D_; g.B = wqkvT; g.ldb = D_; g.K = D_;
    g.Cf = qkvf; g.ldc = NQKV_;
    gemm128<EPI_QKV><<<dim3(S_ / 128, NQKV_ / 128, 1), 256, 0, stream>>>(g);

    rope_kernel<<<S_, 256, 0, stream>>>(qkvf, pos, qb, kb, vtb);

    // ---- attention: scores (scaled, causal+mask), softmax, P.V ----
    reset();
    g.A = qb;  g.lda = D_;          g.batchA = HD_;
    g.B = kb;  g.ldb = KVH_ * HD_;  g.batchB = HD_;  g.bdivB = H_ / KVH_;
    g.K = HD_;
    g.Cf = scores; g.ldc = S_; g.batchC = (long)S_ * S_;
    g.amask = amask; g.scale = 0.08838834764831845f; // 1/sqrt(128)
    gemm128<EPI_SCORES><<<dim3(S_ / 128, S_ / 128, H_), 256, 0, stream>>>(g);

    softmax_kernel<<<dim3(S_, H_), 256, 0, stream>>>(scores, pb);

    reset();
    g.A = pb;  g.lda = S_; g.batchA = (long)S_ * S_;
    g.B = vtb; g.ldb = S_; g.batchB = (long)HD_ * S_; g.bdivB = H_ / KVH_;
    g.K = S_;
    g.Cb = attnb; g.ldc = D_; g.batchC = HD_;
    gemm128<EPI_BF16><<<dim3(S_ / 128, HD_ / 128, H_), 256, 0, stream>>>(g);

    // ---- out proj + residual (writes resid into d_out) ----
    reset();
    g.A = attnb; g.lda = D_; g.B = woutT; g.ldb = D_; g.K = D_;
    g.Cf = out; g.ldc = D_; g.add = x;
    gemm128<EPI_ADDRES><<<dim3(S_ / 128, D_ / 128, 1), 256, 0, stream>>>(g);

    // ---- LN2 + router (fp32 logits) + top-2 routing ----
    (void)hipMemsetAsync(cnt, 0, E_ * sizeof(int), stream);
    ln2_router_kernel<<<S_, 256, 0, stream>>>(out, n2w, wrout, h2b, cnt, tlist,
                                              tok_e, tok_sl, tok_w);

    // ---- MoE: fused SwiGLU up, down proj, gated gather ----
    moe1_kernel<<<dim3(S_ / 128, F_ / 128, E_), 256, 0, stream>>>(h2b, w1b, v1b,
                                                                  tlist, cnt, midb);
    reset();
    g.A = midb; g.lda = F_; g.batchA = (long)S_ * F_;
    g.B = w2T;  g.ldb = F_; g.batchB = (long)D_ * F_;
    g.K = F_;
    g.Cf = ybuf; g.ldc = D_; g.batchC = (long)S_ * D_;
    g.row_cnt = cnt;
    gemm128<EPI_F32><<<dim3(S_ / 128, D_ / 128, E_), 256, 0, stream>>>(g);

    moe_final_kernel<<<dim3(S_, D_ / 256), 256, 0, stream>>>(out, ybuf, tok_e, tok_sl, tok_w);
}
